// TopologicalEncoder_45818711113816
// MI455X (gfx1250) — matrix operand order repaired and checked by reading, not compiled, not run on hardware
//
#include <hip/hip_runtime.h>
#include <hip/hip_bf16.h>
#include <math.h>

// ---------------------------------------------------------------------------
// TopologicalEncoder for MI455X (gfx1250, wave32, WMMA)
// B=32, T=4096, D_IN=256, HID=512, K_LIFT=64, D_MODEL=1024, K_SEL=32
// ---------------------------------------------------------------------------

typedef __attribute__((ext_vector_type(16))) __bf16 v16bf;
typedef __attribute__((ext_vector_type(8)))  float  v8f;

#define BB      32
#define TT      4096
#define DIN     256
#define HID     512
#define KLIFT   64
#define DMODEL  1024
#define KSEL    32
#define AD      259          // DIN + 3
#define BT      (BB * TT)    // 131072 rows

__device__ __forceinline__ unsigned short f2bf(float f) {
    union { float f; unsigned u; } c; c.f = f;
    unsigned u = c.u + 0x7FFFu + ((c.u >> 16) & 1u);   // round-to-nearest-even
    return (unsigned short)(u >> 16);
}

// ---------------------------------------------------------------------------
// K0: convert W1 [256][512] f32 -> bf16, pre-packed in WMMA B-fragment order:
//   out[(((nt*8+ks)*32 + lane)*16) + e], N = nt*16 + (lane&15),
//   K = ks*32 + (lane<16 ? 0 : 8) + (e<8 ? e : e+8)     (ISA 16-bit B 32x16 layout)
// ---------------------------------------------------------------------------
__global__ __launch_bounds__(256) void k0_pack_w1(const float* __restrict__ W1,
                                                  unsigned short* __restrict__ w1b) {
    int idx  = blockIdx.x * 256 + threadIdx.x;     // 0 .. 131071
    int e    = idx & 15;
    int lane = (idx >> 4) & 31;
    int kt   = idx >> 9;                           // nt*8 + ks
    int nt   = kt >> 3, ks = kt & 7;
    int n    = nt * 16 + (lane & 15);
    int k    = ks * 32 + ((lane < 16) ? 0 : 8) + ((e < 8) ? e : e + 8);
    w1b[idx] = f2bf(W1[k * HID + n]);
}

// ---------------------------------------------------------------------------
// K1: fused GEMM  h = tanh(x @ W1 + b1)  reduced against w_sal ->
//     sal = softplus(h . w_sal + b_sal).  h is never materialized.
// Block = 256 thr (8 waves); each wave owns a 16-row strip, sweeps all 512
// cols: 32 n-tiles x 8 k-steps = 256 v_wmma_f32_16x16x32_bf16 per wave.
// ---------------------------------------------------------------------------
__global__ __launch_bounds__(256) void k1_gemm_sal(const float* __restrict__ x,
                                                   const unsigned short* __restrict__ w1b,
                                                   const float* __restrict__ b1,
                                                   const float* __restrict__ w_sal,
                                                   const float* __restrict__ b_sal,
                                                   float* __restrict__ sal) {
    const int tid  = threadIdx.x;
    const int lane = tid & 31;
    const int wid  = tid >> 5;
    const int row_base = blockIdx.x * 128 + wid * 16;
    const int m     = lane & 15;
    const int khalf = (lane < 16) ? 0 : 8;          // ISA 16-bit A 16x32 layout
    const float* arow = x + (size_t)(row_base + m) * DIN;

    // Preload the 8 A-fragments (covers K = 0..255) and convert to bf16.
    union AF { v16bf v; unsigned short s[16]; };
    AF afrag[8];
#pragma unroll
    for (int ks = 0; ks < 8; ++ks) {
        const int k0 = ks * 32 + khalf;
#pragma unroll
        for (int e = 0; e < 8; ++e) {
            afrag[ks].s[e]     = f2bf(arow[k0 + e]);          // K = k0 + e
            afrag[ks].s[8 + e] = f2bf(arow[k0 + 16 + e]);     // K = k0 + 16 + e
        }
    }

    float salacc[8];
#pragma unroll
    for (int j = 0; j < 8; ++j) salacc[j] = 0.f;

    for (int nt = 0; nt < 32; ++nt) {
        v8f c = {0.f, 0.f, 0.f, 0.f, 0.f, 0.f, 0.f, 0.f};
#pragma unroll
        for (int ks = 0; ks < 8; ++ks) {
            // Fragment-shaped bf16 stream: one contiguous 32B load per lane.
            union BF { v16bf v; uint4 q[2]; } bfr;
            const uint4* src =
                reinterpret_cast<const uint4*>(w1b + (((nt * 8 + ks) * 32 + lane) << 4));
            bfr.q[0] = src[0];
            bfr.q[1] = src[1];
            c = __builtin_amdgcn_wmma_f32_16x16x32_bf16(
                    false, afrag[ks].v, false, bfr.v, (short)0, c, false, false);
        }
        const int   n    = nt * 16 + m;
        const float bias = b1[n];
        const float ws   = w_sal[n];
#pragma unroll
        for (int j = 0; j < 8; ++j) {
            float t = tanhf(c[j] + bias);       // row M = j + 8*(lane>=16), col n
            salacc[j] += t * ws;
        }
    }

    // Reduce the 16 N-lanes of each half-wave (masks < 16 keep halves apart).
#pragma unroll
    for (int j = 0; j < 8; ++j) {
        float v = salacc[j];
        v += __shfl_xor(v, 1);
        v += __shfl_xor(v, 2);
        v += __shfl_xor(v, 4);
        v += __shfl_xor(v, 8);
        salacc[j] = v;
    }
    if (m == 0) {
        const float bs = b_sal[0];
        const int rbase = row_base + ((lane >> 4) << 3);
#pragma unroll
        for (int j = 0; j < 8; ++j) {
            float z  = salacc[j] + bs;
            float sp = (z > 20.f) ? z : log1pf(__expf(z));   // stable softplus
            sal[rbase + j] = sp;
        }
    }
}

// ---------------------------------------------------------------------------
// K2: per-batch means needed for centering: mean(x) over T, mean(sal),
//     mean(dsal) = (sal[T-1]-sal[0])/T.   One block per batch row.
// ---------------------------------------------------------------------------
__global__ __launch_bounds__(256) void k2_means(const float* __restrict__ x,
                                                const float* __restrict__ sal,
                                                float* __restrict__ mean_x,
                                                float* __restrict__ mean_sal,
                                                float* __restrict__ dmean) {
    const int b = blockIdx.x, tid = threadIdx.x;
    const float* xb = x + (size_t)b * TT * DIN;
    float s = 0.f;
    for (int t = 0; t < TT; ++t) s += xb[(size_t)t * DIN + tid];   // coalesced across tid
    mean_x[b * DIN + tid] = s * (1.f / TT);

    const float* sb = sal + b * TT;
    float ss = 0.f;
#pragma unroll
    for (int i = 0; i < 16; ++i) ss += sb[tid + i * 256];
    __shared__ float red[8];
    const int lane = tid & 31, wid = tid >> 5;
    for (int msk = 16; msk >= 1; msk >>= 1) ss += __shfl_xor(ss, msk);
    if (lane == 0) red[wid] = ss;
    __syncthreads();
    if (tid == 0) {
        float tot = 0.f;
        for (int i = 0; i < 8; ++i) tot += red[i];
        mean_sal[b] = tot * (1.f / TT);
        dmean[b]    = (sb[TT - 1] - sb[0]) * (1.f / TT);
    }
}

// ---------------------------------------------------------------------------
// K3: relaxed-selector PGD (60 steps, 30-step bisection projection).
// One block per batch row; y lives in LDS; reductions via shfl + 8 LDS slots.
// ---------------------------------------------------------------------------
__global__ __launch_bounds__(256) void k3_selector(const float* __restrict__ sal,
                                                   float* __restrict__ ystar) {
    __shared__ float yb[TT];
    __shared__ float red[8];
    const int b = blockIdx.x, tid = threadIdx.x;
    const int lane = tid & 31, wid = tid >> 5;
    const float* sb = sal + b * TT;

    float sreg[16];
#pragma unroll
    for (int i = 0; i < 16; ++i) {
        const int t = tid + i * 256;
        sreg[i] = sb[t];
        yb[t]   = (float)KSEL / (float)TT;     // y0 = K/T
    }
    __syncthreads();

    for (int step = 0; step < 60; ++step) {
        float yn[16];
        float lmax = 0.f, lsum = 0.f;
#pragma unroll
        for (int i = 0; i < 16; ++i) {
            const int t = tid + i * 256;
            float nb = 0.f;
            if (t >= 1)       nb += yb[t - 1];
            if (t >= 2)       nb += yb[t - 2];
            if (t <= TT - 2)  nb += yb[t + 1];
            if (t <= TT - 3)  nb += yb[t + 2];
            const float y = yb[t];
            const float g = sreg[i] - y - 0.5f * nb;      // 2*lam*y + lam*N y, lam=0.5
            float v = y + 0.1f * g;                        // eta = 0.1
            v = fminf(fmaxf(v, 0.f), 1.f);                 // clip to [0,1]
            yn[i] = v; lsum += v; lmax = fmaxf(lmax, v);
        }
        for (int msk = 16; msk >= 1; msk >>= 1) {
            lmax = fmaxf(lmax, __shfl_xor(lmax, msk));
            lsum += __shfl_xor(lsum, msk);
        }
        __syncthreads();
        if (lane == 0) red[wid] = lmax;
        __syncthreads();
        float hi = red[0];
        for (int i = 1; i < 8; ++i) hi = fmaxf(hi, red[i]);
        __syncthreads();
        if (lane == 0) red[wid] = lsum;
        __syncthreads();
        float sumy = 0.f;
        for (int i = 0; i < 8; ++i) sumy += red[i];

        float lo = 0.f;
        for (int it = 0; it < 30; ++it) {
            const float mid = 0.5f * (lo + hi);
            float ls = 0.f;
#pragma unroll
            for (int i = 0; i < 16; ++i) {
                float v = yn[i] - mid;
                ls += fminf(fmaxf(v, 0.f), 1.f);
            }
            for (int msk = 16; msk >= 1; msk >>= 1) ls += __shfl_xor(ls, msk);
            __syncthreads();
            if (lane == 0) red[wid] = ls;
            __syncthreads();
            float sc = 0.f;
            for (int i = 0; i < 8; ++i) sc += red[i];
            if (sc > (float)KSEL) lo = mid; else hi = mid;   // identical on all threads
        }
        const float tau = (sumy > (float)KSEL) ? 0.5f * (lo + hi) : 0.f;
#pragma unroll
        for (int i = 0; i < 16; ++i) {
            const int t = tid + i * 256;
            yb[t] = fminf(fmaxf(yn[i] - tau, 0.f), 1.f);
        }
        __syncthreads();
    }
#pragma unroll
    for (int i = 0; i < 16; ++i) {
        const int t = tid + i * 256;
        ystar[b * TT + t] = yb[t];
    }
}

// ---------------------------------------------------------------------------
// K4: top-32 indices per row (descending value, lower index on ties),
// iterative argmax with masking in LDS.
// ---------------------------------------------------------------------------
__global__ __launch_bounds__(256) void k4_topk(const float* __restrict__ ystar,
                                               int* __restrict__ topidx) {
    __shared__ float vals[TT];
    __shared__ float rv[8];
    __shared__ int   ri[8];
    const int b = blockIdx.x, tid = threadIdx.x;
    const int lane = tid & 31, wid = tid >> 5;
#pragma unroll
    for (int i = 0; i < 16; ++i) {
        const int t = tid + i * 256;
        vals[t] = ystar[b * TT + t];
    }
    __syncthreads();
    for (int sel = 0; sel < KSEL; ++sel) {
        float bv = -1e30f; int bi = 0x7FFFFFFF;
#pragma unroll
        for (int i = 0; i < 16; ++i) {
            const int t = tid + i * 256;
            const float v = vals[t];
            if (v > bv || (v == bv && t < bi)) { bv = v; bi = t; }
        }
        for (int msk = 16; msk >= 1; msk >>= 1) {
            const float ov = __shfl_xor(bv, msk);
            const int   oi = __shfl_xor(bi, msk);
            if (ov > bv || (ov == bv && oi < bi)) { bv = ov; bi = oi; }
        }
        __syncthreads();
        if (lane == 0) { rv[wid] = bv; ri[wid] = bi; }
        __syncthreads();
        float gv = rv[0]; int gi = ri[0];
        for (int i = 1; i < 8; ++i)
            if (rv[i] > gv || (rv[i] == gv && ri[i] < gi)) { gv = rv[i]; gi = ri[i]; }
        if (tid == 0) { topidx[b * KSEL + sel] = gi; vals[gi] = -1e30f; }
        __syncthreads();
    }
}

// ---------------------------------------------------------------------------
// K5: gather anchors, center/normalize, lift (259->64), project (64->1024).
// Only the 32 selected rows per batch are ever materialized.
// ---------------------------------------------------------------------------
__global__ __launch_bounds__(256) void k5_tokens(const float* __restrict__ x,
                                                 const float* __restrict__ sal,
                                                 const float* __restrict__ mean_x,
                                                 const float* __restrict__ mean_sal,
                                                 const float* __restrict__ dmean,
                                                 const float* __restrict__ mu,
                                                 const float* __restrict__ sigma,
                                                 const float* __restrict__ W_lift,
                                                 const float* __restrict__ b_lift,
                                                 const float* __restrict__ W_proj,
                                                 const float* __restrict__ b_proj,
                                                 const int* __restrict__ topidx,
                                                 float* __restrict__ out) {
    __shared__ float vn[KSEL][AD];      // normalized anchor rows
    __shared__ float lift[KSEL][KLIFT]; // lifted cloud
    __shared__ int   tix[KSEL];
    const int b = blockIdx.x, tid = threadIdx.x;
    if (tid < KSEL) tix[tid] = topidx[b * KSEL + tid];
    __syncthreads();

    for (int idx = tid; idx < KSEL * AD; idx += 256) {
        const int j = idx / AD, d = idx - j * AD;
        const int t = tix[j];
        float val;
        if (d < DIN)        val = x[((size_t)(b * TT + t)) * DIN + d] - mean_x[b * DIN + d];
        else if (d == 256)  val = sal[b * TT + t] - mean_sal[b];
        else if (d == 257)  val = (float)t * (1.f / TT) - ((float)(TT - 1) / (2.f * TT));
        else {
            const float ds = (t == 0) ? 0.f : (sal[b * TT + t] - sal[b * TT + t - 1]);
            val = ds - dmean[b];
        }
        vn[j][d] = (val - mu[d]) / sigma[d];
    }
    __syncthreads();

    for (int idx = tid; idx < KSEL * KLIFT; idx += 256) {
        const int j = idx >> 6, k = idx & 63;
        float acc = b_lift[k];
        for (int d = 0; d < AD; ++d) acc += vn[j][d] * W_lift[d * KLIFT + k];
        lift[j][k] = acc;
    }
    __syncthreads();

    for (int idx = tid; idx < KSEL * DMODEL; idx += 256) {
        const int j = idx >> 10, n = idx & 1023;
        float acc = b_proj[n];
#pragma unroll 8
        for (int k = 0; k < KLIFT; ++k) acc += lift[j][k] * W_proj[k * DMODEL + n];
        out[((size_t)(b * KSEL + j)) * DMODEL + n] = acc;
    }
}

// ---------------------------------------------------------------------------
// Host-side launch
// ---------------------------------------------------------------------------
extern "C" void kernel_launch(void* const* d_in, const int* in_sizes, int n_in,
                              void* d_out, int out_size, void* d_ws, size_t ws_size,
                              hipStream_t stream) {
    const float* x      = (const float*)d_in[0];
    const float* W1     = (const float*)d_in[1];
    const float* b1     = (const float*)d_in[2];
    // d_in[3] = w_event, d_in[4] = b_event  (event_scores not returned -> unused)
    const float* w_sal  = (const float*)d_in[5];
    const float* b_sal  = (const float*)d_in[6];
    const float* mu     = (const float*)d_in[7];
    const float* sigma  = (const float*)d_in[8];
    const float* W_lift = (const float*)d_in[9];
    const float* b_lift = (const float*)d_in[10];
    const float* W_proj = (const float*)d_in[11];
    const float* b_proj = (const float*)d_in[12];

    char* ws = (char*)d_ws;
    unsigned short* w1b      = (unsigned short*)(ws);            // 262144 B
    float*          sal      = (float*)(ws + 262144);            // 524288 B
    float*          mean_x   = (float*)(ws + 786432);            //  32768 B
    float*          mean_sal = (float*)(ws + 819200);            //    128 B
    float*          dmean    = (float*)(ws + 819328);            //    128 B
    int*            topidx   = (int*)  (ws + 819456);            //   4096 B

    float* tokens = (float*)d_out;                               // [32,32,1024]
    float* ystar  = tokens + (size_t)BB * KSEL * DMODEL;         // [32,4096]

    k0_pack_w1 <<<dim3(512),  dim3(256), 0, stream>>>(W1, w1b);
    k1_gemm_sal<<<dim3(1024), dim3(256), 0, stream>>>(x, w1b, b1, w_sal, b_sal, sal);
    k2_means   <<<dim3(BB),   dim3(256), 0, stream>>>(x, sal, mean_x, mean_sal, dmean);
    k3_selector<<<dim3(BB),   dim3(256), 0, stream>>>(sal, ystar);
    k4_topk    <<<dim3(BB),   dim3(256), 0, stream>>>(ystar, topidx);
    k5_tokens  <<<dim3(BB),   dim3(256), 0, stream>>>(x, sal, mean_x, mean_sal, dmean,
                                                      mu, sigma, W_lift, b_lift,
                                                      W_proj, b_proj, topidx, tokens);
}
